// Block_47425028882762
// MI455X (gfx1250) — matrix operand order repaired and checked by reading, not compiled, not run on hardware
//
#include <hip/hip_runtime.h>
#include <hip/hip_bf16.h>
#include <cstdint>

// ---------------- model constants ----------------
#define DD     1024
#define HH     8
#define QLR    384
#define KVLR   256
#define NOPE   128
#define ROPEW  64
#define VD     128
#define NEXP   8
#define INTER  512
#define BB     2
#define TT     2048
#define QKD    192          // NOPE + ROPEW
#define KVD    256          // NOPE + VD
#define NTOK   (BB*TT)      // 4096
#define FSCALE 0.07216878364870323f   // 1/sqrt(192)

typedef __attribute__((ext_vector_type(16))) __bf16 v16bf;
typedef __attribute__((ext_vector_type(8)))  float  v8f;
typedef unsigned short u16;
typedef unsigned int   u32;

// ---------------- small helpers ----------------
__device__ __forceinline__ u16 f32_to_bf16(float f) {
  u32 u = __builtin_bit_cast(u32, f);
  u32 r = u + 0x7FFFu + ((u >> 16) & 1u);   // round-to-nearest-even
  return (u16)(r >> 16);
}
__device__ __forceinline__ float bf16_to_f32(u16 h) {
  u32 u = ((u32)h) << 16;
  return __builtin_bit_cast(float, u);
}
__device__ __forceinline__ v8f vzero() {
  v8f z = {0.f,0.f,0.f,0.f,0.f,0.f,0.f,0.f};
  return z;
}
__device__ __forceinline__ v8f wmma_bf16(const u32 a[8], const u32 b[8], v8f c) {
  union U { u32 u[8]; v16bf v; } ua, ub;
#pragma unroll
  for (int i = 0; i < 8; ++i) { ua.u[i] = a[i]; ub.u[i] = b[i]; }
  // D = A(16x32 bf16) * B(32x16 bf16) + C(16x16 f32)
  return __builtin_amdgcn_wmma_f32_16x16x32_bf16(false, ua.v, false, ub.v,
                                                 (short)0, c, false, false);
}
// wave-local LDS fence (flash attention waves have divergent loop counts, so a
// workgroup barrier is unusable there; its LDS regions are wave-private).
#define WAVE_LDS_FENCE() asm volatile("s_wait_dscnt 0" ::: "memory")

// CDNA5 async memory->LDS copy (tracked with ASYNCcnt, not LOADcnt):
// each lane moves 16B from its global address to its LDS address.
__device__ __forceinline__ void async_load_b128(u32 lds_off, const void* gptr) {
  asm volatile("global_load_async_to_lds_b128 %0, %1, off"
               :: "v"(lds_off), "v"(gptr) : "memory");
}
#define WAIT_ASYNC_LE3() asm volatile("s_wait_asynccnt 0x3" ::: "memory")
#define WAIT_ASYNC_0()   asm volatile("s_wait_asynccnt 0x0" ::: "memory")

// ---------------- elementwise kernels ----------------
__global__ void convert_f32_bf16_kernel(const float* __restrict__ in,
                                        u16* __restrict__ out, int n) {
  for (int i = blockIdx.x * blockDim.x + threadIdx.x; i < n;
       i += gridDim.x * blockDim.x)
    out[i] = f32_to_bf16(in[i]);
}

// w[K][N] (f32) -> wT[N][K] (bf16), batched over blockIdx.z
__global__ __launch_bounds__(256)
void convt_bf16_kernel(const float* __restrict__ w, u16* __restrict__ wT,
                       int K, int N) {
  __shared__ float tile[32][33];
  const float* wz = w + (size_t)blockIdx.z * K * N;
  u16* wTz = wT + (size_t)blockIdx.z * K * N;
  const int k0 = blockIdx.x * 32, n0 = blockIdx.y * 32;
  const int tx = threadIdx.x & 31, ty = threadIdx.x >> 5;   // 32 x 8
#pragma unroll
  for (int i = 0; i < 4; ++i) {
    int k = ty + i * 8;
    tile[k][tx] = wz[(size_t)(k0 + k) * N + n0 + tx];
  }
  __syncthreads();
#pragma unroll
  for (int i = 0; i < 4; ++i) {
    int n = ty + i * 8;
    wTz[(size_t)(n0 + n) * K + k0 + tx] = f32_to_bf16(tile[tx][n]);
  }
}

__global__ __launch_bounds__(256)
void rms_bf16_kernel(const float* __restrict__ in, int ld,
                     const float* __restrict__ w,
                     u16* __restrict__ out, int len) {
  __shared__ float red[8];
  const int row = blockIdx.x;
  const float* r = in + (size_t)row * ld;
  float ss = 0.f;
  for (int i = threadIdx.x; i < len; i += 256) { float v = r[i]; ss += v * v; }
#pragma unroll
  for (int o = 16; o; o >>= 1) ss += __shfl_xor(ss, o, 32);
  if ((threadIdx.x & 31) == 0) red[threadIdx.x >> 5] = ss;
  __syncthreads();
  if (threadIdx.x == 0) {
    float t = 0.f;
    for (int i = 0; i < 8; ++i) t += red[i];
    red[0] = rsqrtf(t / (float)len + 1e-6f);
  }
  __syncthreads();
  const float sc = red[0];
  for (int i = threadIdx.x; i < len; i += 256)
    out[(size_t)row * len + i] = f32_to_bf16(r[i] * sc * w[i]);
}

__global__ void silu_mul_kernel(const float* __restrict__ a,
                                const float* __restrict__ b,
                                u16* __restrict__ out, int n) {
  for (int i = blockIdx.x * blockDim.x + threadIdx.x; i < n;
       i += gridDim.x * blockDim.x) {
    float x = a[i];
    float s = x / (1.f + __expf(-x));
    out[i] = f32_to_bf16(s * b[i]);
  }
}

__global__ void add2_kernel(const float* __restrict__ a,
                            const float* __restrict__ b,
                            float* __restrict__ out, int n) {
  for (int i = blockIdx.x * blockDim.x + threadIdx.x; i < n;
       i += gridDim.x * blockDim.x)
    out[i] = a[i] + b[i];
}

// q (f32, [NTOK, H*QKD]) -> bf16 with RoPE applied to last 64 dims of each head
__global__ void build_qb_kernel(const float* __restrict__ q,
                                const float* __restrict__ cosT,
                                const float* __restrict__ sinT,
                                u16* __restrict__ qb) {
  const int total = NTOK * HH * QKD;
  for (int idx = blockIdx.x * blockDim.x + threadIdx.x; idx < total;
       idx += gridDim.x * blockDim.x) {
    int n = idx / (HH * QKD);
    int rem = idx - n * (HH * QKD);
    int h = rem / QKD;
    int d = rem - h * QKD;
    float v;
    if (d < NOPE) {
      v = q[idx];
    } else {
      int pos = d - NOPE, i = pos >> 1, t = n % TT;
      float c = cosT[t * (ROPEW / 2) + i], s = sinT[t * (ROPEW / 2) + i];
      float x0 = q[n * (HH * QKD) + h * QKD + NOPE + 2 * i];
      float x1 = q[n * (HH * QKD) + h * QKD + NOPE + 2 * i + 1];
      v = (pos & 1) ? (x0 * s + x1 * c) : (x0 * c - x1 * s);
    }
    qb[idx] = f32_to_bf16(v);
  }
}

// k (bf16, [NTOK, H*QKD]): nope part from kv, rope part from lat k_r slice
__global__ void build_kb_kernel(const float* __restrict__ kv,
                                const float* __restrict__ lat,
                                const float* __restrict__ cosT,
                                const float* __restrict__ sinT,
                                u16* __restrict__ kb) {
  const int total = NTOK * HH * QKD;
  for (int idx = blockIdx.x * blockDim.x + threadIdx.x; idx < total;
       idx += gridDim.x * blockDim.x) {
    int n = idx / (HH * QKD);
    int rem = idx - n * (HH * QKD);
    int h = rem / QKD;
    int d = rem - h * QKD;
    float v;
    if (d < NOPE) {
      v = kv[n * (HH * KVD) + h * KVD + d];
    } else {
      int pos = d - NOPE, i = pos >> 1, t = n % TT;
      float c = cosT[t * (ROPEW / 2) + i], s = sinT[t * (ROPEW / 2) + i];
      float x0 = lat[n * (QLR + KVLR + ROPEW) + QLR + KVLR + 2 * i];
      float x1 = lat[n * (QLR + KVLR + ROPEW) + QLR + KVLR + 2 * i + 1];
      v = (pos & 1) ? (x0 * s + x1 * c) : (x0 * c - x1 * s);
    }
    kb[idx] = f32_to_bf16(v);
  }
}

// v transposed: vbT[(h*VD + vc)*NTOK + n]  (token-minor -> contiguous K-pairs
// for WMMA B-fragments in the P@V stage)
__global__ void build_vT_kernel(const float* __restrict__ kv,
                                u16* __restrict__ vbT) {
  const int total = NTOK * HH * VD;
  for (int idx = blockIdx.x * blockDim.x + threadIdx.x; idx < total;
       idx += gridDim.x * blockDim.x) {
    int n = idx / (HH * VD);
    int rem = idx - n * (HH * VD);
    int h = rem / VD;
    int vc = rem - h * VD;
    vbT[(size_t)(h * VD + vc) * NTOK + n] =
        f32_to_bf16(kv[n * (HH * KVD) + h * KVD + NOPE + vc]);
  }
}

// gate: logits = h2 @ gate_w, softmax over 8, keep top-2 probs (rest 0)
__global__ __launch_bounds__(32)
void gate_topk_kernel(const u16* __restrict__ h2b,
                      const float* __restrict__ gw,
                      float* __restrict__ cw) {
  const int n = blockIdx.x;
  const int lane = threadIdx.x;
  float acc[NEXP];
#pragma unroll
  for (int e = 0; e < NEXP; ++e) acc[e] = 0.f;
  for (int d = lane; d < DD; d += 32) {
    float hv = bf16_to_f32(h2b[(size_t)n * DD + d]);
#pragma unroll
    for (int e = 0; e < NEXP; ++e) acc[e] += hv * gw[d * NEXP + e];
  }
#pragma unroll
  for (int e = 0; e < NEXP; ++e)
#pragma unroll
    for (int o = 16; o; o >>= 1) acc[e] += __shfl_xor(acc[e], o, 32);
  if (lane == 0) {
    float mx = acc[0];
#pragma unroll
    for (int e = 1; e < NEXP; ++e) mx = fmaxf(mx, acc[e]);
    float p[NEXP], sum = 0.f;
#pragma unroll
    for (int e = 0; e < NEXP; ++e) { p[e] = __expf(acc[e] - mx); sum += p[e]; }
    float inv = 1.f / sum;
#pragma unroll
    for (int e = 0; e < NEXP; ++e) p[e] *= inv;
    int i0 = 0;
#pragma unroll
    for (int e = 1; e < NEXP; ++e) if (p[e] > p[i0]) i0 = e;
    int i1 = (i0 == 0) ? 1 : 0;
#pragma unroll
    for (int e = 0; e < NEXP; ++e)
      if (e != i0 && p[e] > p[i1]) i1 = e;
#pragma unroll
    for (int e = 0; e < NEXP; ++e)
      cw[(size_t)n * NEXP + e] = (e == i0 || e == i1) ? p[e] : 0.f;
  }
}

// ---------------- WMMA GEMM (async-to-LDS, double buffered) ----------------
// C[M,N](f32) = A[M,K](bf16) @ Bt[N,K](bf16, pre-transposed)  (+res)(*rowscale)
// Block: 256 threads = 8 waves; tile BM=128, BN=64, BK=32.
// Waves 4(m) x 2(n): each wave owns a 32x32 sub-tile = 2x2 WMMA accumulators.
// Tiles staged with GLOBAL_LOAD_ASYNC_TO_LDS_B128 (3 async ops / wave / step),
// double buffered: issue(next) -> s_wait_asynccnt<=3 -> barrier -> compute.
__global__ __launch_bounds__(256)
void gemm_bf16_kernel(const u16* __restrict__ A, const u16* __restrict__ Bt,
                      float* __restrict__ C, const float* __restrict__ res,
                      const float* __restrict__ rowscale, int rs_stride,
                      int M, int N, int K, int accumulate) {
  __shared__ __align__(16) u16 sA[2][128 * 40];   // [row][k], stride 40
  __shared__ __align__(16) u16 sB[2][64 * 40];    // [n][k],   stride 40
  const int tid  = threadIdx.x;
  const int lane = tid & 31;
  const int wave = tid >> 5;
  const int wm = wave >> 1, wn = wave & 1;
  const int m0 = blockIdx.x * 128;
  const int n0 = blockIdx.y * 64;
  const int l15 = lane & 15;
  const int hi  = (lane >= 16);

  // per-thread 16B-chunk coordinates (row = tid/4, chunk col = tid%4)
  const int crow = tid >> 2, cc = (tid & 3) * 8;
  u32 aL[2], bL[2];
#pragma unroll
  for (int i = 0; i < 2; ++i) {
    aL[i] = (u32)(uintptr_t)&sA[i][crow * 40 + cc];
    bL[i] = (u32)(uintptr_t)&sB[i][crow * 40 + cc];
  }
  const u16* gA  = A  + (size_t)(m0 + crow) * K + cc;   // rows 0..63
  const u16* gA2 = gA + (size_t)64 * K;                 // rows 64..127
  const u16* gB  = Bt + (size_t)(n0 + crow) * K + cc;   // crow in 0..63

  v8f acc[2][2];
#pragma unroll
  for (int a = 0; a < 2; ++a)
#pragma unroll
    for (int b = 0; b < 2; ++b) acc[a][b] = vzero();

  auto issue = [&](int buf, int k0) {
    async_load_b128(aL[buf],                     gA  + k0);
    async_load_b128(aL[buf] + 64 * 40 * 2,       gA2 + k0);
    async_load_b128(bL[buf],                     gB  + k0);
  };

  const int nk = K >> 5;
  issue(0, 0);
  for (int it = 0; it < nk; ++it) {
    const int buf = it & 1;
    if (it + 1 < nk) {
      issue(buf ^ 1, (it + 1) << 5);
      WAIT_ASYNC_LE3();        // in-order completion => current tile landed
    } else {
      WAIT_ASYNC_0();
    }
    __syncthreads();           // all waves' tile data visible

    // gather fragments per CDNA5 16-bit WMMA layouts
    u32 af[2][8], bfr[2][8];
#pragma unroll
    for (int tm = 0; tm < 2; ++tm) {
      int row = wm * 32 + tm * 16 + l15;
#pragma unroll
      for (int v = 0; v < 8; ++v) {
        int k = 2 * (v & 3) + ((v >= 4) ? 16 : 0) + (hi ? 8 : 0);
        af[tm][v] = *reinterpret_cast<const u32*>(&sA[buf][row * 40 + k]);
      }
    }
#pragma unroll
    for (int tn = 0; tn < 2; ++tn) {
      int col = wn * 32 + tn * 16 + l15;
#pragma unroll
      for (int v = 0; v < 8; ++v) {
        int k = 2 * (v & 3) + ((v >= 4) ? 16 : 0) + (hi ? 8 : 0);
        bfr[tn][v] = *reinterpret_cast<const u32*>(&sB[buf][col * 40 + k]);
      }
    }
#pragma unroll
    for (int tm = 0; tm < 2; ++tm)
#pragma unroll
      for (int tn = 0; tn < 2; ++tn)
        acc[tm][tn] = wmma_bf16(af[tm], bfr[tn], acc[tm][tn]);
    __syncthreads();           // done reading buf before it is re-targeted
  }

  // epilogue (C layout: VGPR j -> rows j / j+8, N = lane%16)
  const int rofs = hi ? 8 : 0;
#pragma unroll
  for (int tm = 0; tm < 2; ++tm)
#pragma unroll
    for (int tn = 0; tn < 2; ++tn)
#pragma unroll
      for (int j = 0; j < 8; ++j) {
        int row = m0 + wm * 32 + tm * 16 + j + rofs;
        int col = n0 + wn * 32 + tn * 16 + l15;
        float v = acc[tm][tn][j];
        if (rowscale) v *= rowscale[(size_t)row * rs_stride];
        if (res) v += res[(size_t)row * N + col];
        size_t o = (size_t)row * N + col;
        if (accumulate) C[o] += v; else C[o] = v;
      }
}

// ---------------- flash attention ----------------
// Block: 128 threads = 4 waves; each wave owns 16 query rows; 32-key tiles.
// grid = (T/64, H, B). Causal. Online softmax, f32 accumulation, bf16 WMMA.
__global__ __launch_bounds__(128)
void flash_attn_kernel(const u16* __restrict__ qb, const u16* __restrict__ kb,
                       const u16* __restrict__ vbT, float* __restrict__ y) {
  __shared__ float sS[4][16][33];
  __shared__ __align__(4) u16 sP[4][16][40];
  __shared__ float sRow[4][16];

  const int lane = threadIdx.x & 31;
  const int wave = threadIdx.x >> 5;
  const int qt = blockIdx.x, h = blockIdx.y, b = blockIdx.z;
  const int qbase = qt * 64 + wave * 16;
  const int l15 = lane & 15;
  const int hi  = (lane >= 16);
  const int rofs = hi ? 8 : 0;

  // preload Q A-fragments (loop invariant): 6 d-steps of 32
  u32 qf[6][8];
  {
    const u16* base = qb + (size_t)(b * TT + qbase + l15) * (HH * QKD) + h * QKD;
#pragma unroll
    for (int s = 0; s < 6; ++s)
#pragma unroll
      for (int v = 0; v < 8; ++v) {
        int k = 2 * (v & 3) + ((v >= 4) ? 16 : 0) + rofs;
        qf[s][v] = *reinterpret_cast<const u32*>(base + s * 32 + k);
      }
  }

  float m_i = -3e38f, l_i = 0.f;
  v8f o[8];
#pragma unroll
  for (int t = 0; t < 8; ++t) o[t] = vzero();

  const int kmax = qbase + 15;
  for (int kt = 0; kt * 32 <= kmax; ++kt) {
    const int kbase = kt * 32;

    // ---- S = Q @ K^T for 32 keys (two 16-key column tiles) ----
    v8f s0 = vzero(), s1 = vzero();
    {
      const u16* kb0 =
          kb + (size_t)(b * TT + kbase + l15) * (HH * QKD) + h * QKD;
      const u16* kb1 = kb0 + (size_t)16 * (HH * QKD);
#pragma unroll
      for (int s = 0; s < 6; ++s) {
        u32 kf0[8], kf1[8];
#pragma unroll
        for (int v = 0; v < 8; ++v) {
          int k = s * 32 + 2 * (v & 3) + ((v >= 4) ? 16 : 0) + (hi ? 8 : 0);
          kf0[v] = *reinterpret_cast<const u32*>(kb0 + k);
          kf1[v] = *reinterpret_cast<const u32*>(kb1 + k);
        }
        s0 = wmma_bf16(qf[s], kf0, s0);
        s1 = wmma_bf16(qf[s], kf1, s1);
      }
    }

    // ---- scale + causal mask, spill S tile to wave-private LDS ----
#pragma unroll
    for (int j = 0; j < 8; ++j) {
      int r = j + rofs;
      int qtok = qbase + r;
      int ka = kbase + l15;
      float va = s0[j] * FSCALE; if (ka > qtok)       va = -3e38f;
      float vb = s1[j] * FSCALE; if (ka + 16 > qtok)  vb = -3e38f;
      sS[wave][r][l15]      = va;
      sS[wave][r][16 + l15] = vb;
    }
    WAVE_LDS_FENCE();

    // ---- online softmax: lane pair (l, l+16) owns row l, 16 cols each ----
    {
      const int cb = hi ? 16 : 0;
      float pm = -3e38f;
#pragma unroll
      for (int c = 0; c < 16; ++c) pm = fmaxf(pm, sS[wave][l15][cb + c]);
      pm = fmaxf(pm, __shfl_xor(pm, 16, 32));
      float m_new = fmaxf(m_i, pm);
      float alpha = __expf(m_i - m_new);
      float ps = 0.f;
#pragma unroll
      for (int c = 0; c < 16; ++c) {
        float p = __expf(sS[wave][l15][cb + c] - m_new);
        ps += p;
        sP[wave][l15][cb + c] = f32_to_bf16(p);
      }
      ps += __shfl_xor(ps, 16, 32);
      l_i = l_i * alpha + ps;
      m_i = m_new;
      if (lane < 16) sRow[wave][lane] = alpha;
    }
    WAVE_LDS_FENCE();

    // ---- rescale O by alpha(row) ----
    float al[8];
#pragma unroll
    for (int j = 0; j < 8; ++j) al[j] = sRow[wave][j + rofs];
#pragma unroll
    for (int t = 0; t < 8; ++t)
#pragma unroll
      for (int j = 0; j < 8; ++j) o[t][j] *= al[j];

    // ---- O += P @ V (P from LDS as A-frag, V from transposed global) ----
    u32 pf[8];
#pragma unroll
    for (int v = 0; v < 8; ++v) {
      int k = 2 * (v & 3) + ((v >= 4) ? 16 : 0) + rofs;
      pf[v] = *reinterpret_cast<const u32*>(&sP[wave][l15][k]);
    }
#pragma unroll
    for (int t = 0; t < 8; ++t) {
      u32 vf[8];
      int vc = t * 16 + l15;
      const u16* vb0 = vbT + (size_t)(h * VD + vc) * NTOK + b * TT + kbase;
#pragma unroll
      for (int v = 0; v < 8; ++v) {
        int k = 2 * (v & 3) + ((v >= 4) ? 16 : 0) + (hi ? 8 : 0);
        vf[v] = *reinterpret_cast<const u32*>(vb0 + k);
      }
      o[t] = wmma_bf16(pf, vf, o[t]);
    }
  }

  // ---- normalize by row sums and store ----
  if (lane < 16) sRow[wave][lane] = l_i;
  WAVE_LDS_FENCE();
  float ls[8];
#pragma unroll
  for (int j = 0; j < 8; ++j) ls[j] = sRow[wave][j + rofs];
#pragma unroll
  for (int t = 0; t < 8; ++t)
#pragma unroll
    for (int j = 0; j < 8; ++j) {
      int tok = qbase + j + rofs;
      int col = h * VD + t * 16 + l15;
      y[(size_t)(b * TT + tok) * DD + col] = o[t][j] / ls[j];
    }
}

// ---------------- host driver ----------------
extern "C" void kernel_launch(void* const* d_in, const int* in_sizes, int n_in,
                              void* d_out, int out_size, void* d_ws,
                              size_t ws_size, hipStream_t stream) {
  (void)in_sizes; (void)n_in; (void)out_size; (void)ws_size;
  const float* x        = (const float*)d_in[0];
  const float* fcos     = (const float*)d_in[1];
  const float* fsin     = (const float*)d_in[2];
  const float* rmsn1_w  = (const float*)d_in[3];
  const float* rmsn2_w  = (const float*)d_in[4];
  const float* latent_w = (const float*)d_in[5];
  const float* q_norm_w = (const float*)d_in[6];
  const float* q_up_w   = (const float*)d_in[7];
  const float* kv_norm_w= (const float*)d_in[8];
  const float* kv_up_w  = (const float*)d_in[9];
  const float* c_proj_w = (const float*)d_in[10];
  const float* gate_w   = (const float*)d_in[11];
  const float* sh_w1    = (const float*)d_in[12];
  const float* sh_w2    = (const float*)d_in[13];
  const float* sh_w3    = (const float*)d_in[14];
  const float* e_w1     = (const float*)d_in[15];
  const float* e_w2     = (const float*)d_in[16];
  const float* e_w3     = (const float*)d_in[17];

  char* base = (char*)d_ws;
  size_t off = 0;
  auto alloc = [&](size_t bytes) -> void* {
    void* r = base + off;
    off = (off + bytes + 255) & ~(size_t)255;
    return r;
  };

  const int LATW = QLR + KVLR + ROPEW;   // 704
  // bf16 buffers (weights stored TRANSPOSED: [N][K])
  u16* hb      = (u16*)alloc((size_t)NTOK * DD * 2);
  u16* wb_lat  = (u16*)alloc((size_t)DD * LATW * 2);
  u16* wb_qup  = (u16*)alloc((size_t)QLR * HH * QKD * 2);
  u16* wb_kvup = (u16*)alloc((size_t)KVLR * HH * KVD * 2);
  u16* wb_cprj = (u16*)alloc((size_t)DD * DD * 2);
  u16* wb_sh1  = (u16*)alloc((size_t)DD * 1024 * 2);
  u16* wb_sh2  = (u16*)alloc((size_t)1024 * DD * 2);
  u16* wb_sh3  = (u16*)alloc((size_t)DD * 1024 * 2);
  u16* wb_e1   = (u16*)alloc((size_t)NEXP * DD * INTER * 2);
  u16* wb_e2   = (u16*)alloc((size_t)NEXP * INTER * DD * 2);
  u16* wb_e3   = (u16*)alloc((size_t)NEXP * DD * INTER * 2);
  u16* qlb     = (u16*)alloc((size_t)NTOK * QLR * 2);
  u16* kvlb    = (u16*)alloc((size_t)NTOK * KVLR * 2);
  u16* qbb     = (u16*)alloc((size_t)NTOK * HH * QKD * 2);
  u16* kbb     = (u16*)alloc((size_t)NTOK * HH * QKD * 2);
  u16* vbT     = (u16*)alloc((size_t)HH * VD * NTOK * 2);
  u16* h2b     = (u16*)alloc((size_t)NTOK * DD * 2);
  u16* tbb     = (u16*)alloc((size_t)NTOK * 1024 * 2);
  // f32 buffers
  float* lat   = (float*)alloc((size_t)NTOK * LATW * 4);
  float* qf_   = (float*)alloc((size_t)NTOK * HH * QKD * 4);
  float* kvf_  = (float*)alloc((size_t)NTOK * HH * KVD * 4);
  float* yatt  = (float*)alloc((size_t)NTOK * DD * 4);
  float* x2    = (float*)alloc((size_t)NTOK * DD * 4);
  float* ga    = (float*)alloc((size_t)NTOK * 1024 * 4);
  float* gb    = (float*)alloc((size_t)NTOK * 1024 * 4);
  float* ymoe  = (float*)alloc((size_t)NTOK * DD * 4);
  float* cw    = (float*)alloc((size_t)NTOK * NEXP * 4);

  auto conv = [&](const float* in, u16* outp, int n) {
    convert_f32_bf16_kernel<<<(n + 255) / 256, 256, 0, stream>>>(in, outp, n);
  };
  // weight [K][N] f32 -> [N][K] bf16, batched
  auto convT = [&](const float* w, u16* wT, int K, int N, int batch) {
    dim3 g(K / 32, N / 32, batch);
    convt_bf16_kernel<<<g, 256, 0, stream>>>(w, wT, K, N);
  };
  auto gemm = [&](const u16* A, const u16* Bt, float* C, int M, int N, int K,
                  const float* res, const float* rowscale, int rs_stride,
                  int accum) {
    dim3 g(M / 128, N / 64);
    gemm_bf16_kernel<<<g, 256, 0, stream>>>(A, Bt, C, res, rowscale, rs_stride,
                                            M, N, K, accum);
  };

  // 1. rms1 + latent projection
  rms_bf16_kernel<<<NTOK, 256, 0, stream>>>(x, DD, rmsn1_w, hb, DD);
  convT(latent_w, wb_lat, DD, LATW, 1);
  gemm(hb, wb_lat, lat, NTOK, LATW, DD, nullptr, nullptr, 0, 0);

  // 2. q / kv up-projections
  rms_bf16_kernel<<<NTOK, 256, 0, stream>>>(lat, LATW, q_norm_w, qlb, QLR);
  rms_bf16_kernel<<<NTOK, 256, 0, stream>>>(lat + QLR, LATW, kv_norm_w, kvlb,
                                            KVLR);
  convT(q_up_w, wb_qup, QLR, HH * QKD, 1);
  gemm(qlb, wb_qup, qf_, NTOK, HH * QKD, QLR, nullptr, nullptr, 0, 0);
  convT(kv_up_w, wb_kvup, KVLR, HH * KVD, 1);
  gemm(kvlb, wb_kvup, kvf_, NTOK, HH * KVD, KVLR, nullptr, nullptr, 0, 0);

  // 3. RoPE + bf16 packing for attention operands
  {
    int n1 = NTOK * HH * QKD;
    build_qb_kernel<<<(n1 + 255) / 256, 256, 0, stream>>>(qf_, fcos, fsin, qbb);
    build_kb_kernel<<<(n1 + 255) / 256, 256, 0, stream>>>(kvf_, lat, fcos,
                                                          fsin, kbb);
    int n2 = NTOK * HH * VD;
    build_vT_kernel<<<(n2 + 255) / 256, 256, 0, stream>>>(kvf_, vbT);
  }

  // 4. causal flash attention
  flash_attn_kernel<<<dim3(TT / 64, HH, BB), 128, 0, stream>>>(qbb, kbb, vbT,
                                                               yatt);

  // 5. output projection + residual
  conv(yatt, tbb, NTOK * DD);
  convT(c_proj_w, wb_cprj, DD, DD, 1);
  gemm(tbb, wb_cprj, x2, NTOK, DD, DD, x, nullptr, 0, 0);

  // 6. rms2, shared MLP
  rms_bf16_kernel<<<NTOK, 256, 0, stream>>>(x2, DD, rmsn2_w, h2b, DD);
  convT(sh_w1, wb_sh1, DD, 1024, 1);
  convT(sh_w3, wb_sh3, DD, 1024, 1);
  convT(sh_w2, wb_sh2, 1024, DD, 1);
  gemm(h2b, wb_sh1, ga, NTOK, 1024, DD, nullptr, nullptr, 0, 0);
  gemm(h2b, wb_sh3, gb, NTOK, 1024, DD, nullptr, nullptr, 0, 0);
  silu_mul_kernel<<<(NTOK * 1024 + 255) / 256, 256, 0, stream>>>(
      ga, gb, tbb, NTOK * 1024);
  gemm(tbb, wb_sh2, ymoe, NTOK, DD, 1024, nullptr, nullptr, 0, 0);

  // 7. gate + dense experts (gate weight fused into down-proj epilogue)
  gate_topk_kernel<<<NTOK, 32, 0, stream>>>(h2b, gate_w, cw);
  convT(e_w1, wb_e1, DD, INTER, NEXP);
  convT(e_w2, wb_e2, INTER, DD, NEXP);
  convT(e_w3, wb_e3, DD, INTER, NEXP);
  for (int e = 0; e < NEXP; ++e) {
    gemm(h2b, wb_e1 + (size_t)e * DD * INTER, ga, NTOK, INTER, DD, nullptr,
         nullptr, 0, 0);
    gemm(h2b, wb_e3 + (size_t)e * DD * INTER, gb, NTOK, INTER, DD, nullptr,
         nullptr, 0, 0);
    silu_mul_kernel<<<(NTOK * INTER + 255) / 256, 256, 0, stream>>>(
        ga, gb, tbb, NTOK * INTER);
    gemm(tbb, wb_e2 + (size_t)e * INTER * DD, ymoe, NTOK, DD, INTER, nullptr,
         cw + e, NEXP, 1);
  }

  // 8. final residual
  add2_kernel<<<(NTOK * DD + 255) / 256, 256, 0, stream>>>(
      x2, ymoe, (float*)d_out, NTOK * DD);
}
